// OriginalDecoder_15925738733982
// MI455X (gfx1250) — matrix operand order repaired and checked by reading, not compiled, not run on hardware
//
#include <hip/hip_runtime.h>
#include <hip/hip_bf16.h>

// ---------------------------------------------------------------------------
// Show-Attend-Tell decoder for MI455X (gfx1250, wave32, WMMA).
// Precision: bf16 operands + f32 accumulation via v_wmma_f32_16x16x32_bf16.
// Recurrent working set (bf16 weights + att1 + f32 encoder, ~170MB) fits the
// 192MB L2, so the 31-step sequential loop is matrix-pipe/L2-latency bound.
// GEMM k-loop: compile-time trip count, double-buffered fragments, and an
// accumulator-carrying asm barrier: WMMA(i) depends on barrier(i-1) output and
// barrier(i) depends on WMMA(i) output, while the "memory" clobber pins the
// loads. Emitted order is loads(i+1) | wmma(i) | barrier, so waits before the
// WMMA chain are loadcnt<=2+JT, not 0 (one partial wait amortized over JT
// WMMAs instead of a full L2 latency per WMMA).
// Workspace requirement: ~128 MB.
// ---------------------------------------------------------------------------

typedef __bf16 bf16_t;
typedef __attribute__((ext_vector_type(16))) __bf16 v16bf;
typedef __attribute__((ext_vector_type(8)))  __bf16 v8bf;
typedef __attribute__((ext_vector_type(8)))  float  v8f;

#define DEC_B   64
#define DEC_P   196
#define DEC_T   32
#define DEC_ENC 2048
#define DEC_ATT 512
#define DEC_DEC 512
#define DEC_EMB 512
#define DEC_V   30000
#define XDIM    (DEC_EMB + DEC_ENC)   /* 2560 */
#define G3      (3 * DEC_DEC)         /* 1536 */

union AFrag { v16bf v; struct { v8bf lo, hi; } h; };

// Accumulator-carrying scheduling barrier: orders the WMMA chain between
// consecutive barriers and pins memory ops (loads) to their region.
template <int JT>
__device__ __forceinline__ void acc_barrier(v8f* acc)
{
    if constexpr (JT == 4)
        asm volatile("" : "+v"(acc[0]), "+v"(acc[1]), "+v"(acc[2]), "+v"(acc[3]) :: "memory");
    else if constexpr (JT == 3)
        asm volatile("" : "+v"(acc[0]), "+v"(acc[1]), "+v"(acc[2]) :: "memory");
    else if constexpr (JT == 2)
        asm volatile("" : "+v"(acc[0]), "+v"(acc[1]) :: "memory");
    else
        asm volatile("" : "+v"(acc[0]) :: "memory");
}

// ---------------------------------------------------------------------------
// GEMM: C[M, JT*16*gridDim.x] = A[M,K] * W[N,K]^T + bias, all tiles full.
// A, W bf16 row-major (K contiguous); C f32 with row stride ldc.
// Block = 128 threads = 4 waves; wave w computes rows m0=(by*4+w)*16,
// cols [bx*JT*16, (bx+1)*JT*16). JT and K compile-time; K multiple of 32.
// ---------------------------------------------------------------------------
template <int JT, int K>
__global__ __launch_bounds__(128)
void gemm_bf16_wmma(const bf16_t* __restrict__ A, const bf16_t* __restrict__ W,
                    const float* __restrict__ bias, float* __restrict__ C,
                    int M, int ldc)
{
    const int waveId = threadIdx.x >> 5;
    const int lane   = threadIdx.x & 31;
    const int half   = lane >> 4;          // 0: lanes 0-15, 1: lanes 16-31
    const int row16  = lane & 15;          // M-row / N-col within tile

    const int m0 = (blockIdx.y * 4 + waveId) * 16;
    if (m0 >= M) return;                   // wave-uniform exit (EXEC stays full)
    const int nbase = blockIdx.x * (JT * 16);

    const bf16_t* parow = A + (size_t)(m0 + row16) * K + 16 * half;      // A half base
    const bf16_t* pw[JT];                                                // B row bases
#pragma unroll
    for (int j = 0; j < JT; ++j)
        pw[j] = W + (size_t)(nbase + j * 16 + row16) * K + 16 * half;

    // ---- A fragment: 16x32, lane=row; elems 0..15 -> K=k0+16*half+s
    //      (contiguous, matches B): two contiguous b128.
    // ---- B fragment: 32x16, lane=col; 16 contiguous K at k0+16*half: b256.
    AFrag a[2];
    v16bf b[2][JT];
    v8f   acc[JT] = {};

    constexpr int NK = K / 32;

    // prologue: load fragments for k0 = 0 into buffer 0
    a[0].h.lo = *(const v8bf*)(parow);
    a[0].h.hi = *(const v8bf*)(parow + 8);
#pragma unroll
    for (int j = 0; j < JT; ++j) b[0][j] = *(const v16bf*)(pw[j]);
    acc_barrier<JT>(acc);

#pragma unroll 2
    for (int it = 0; it < NK - 1; ++it) {
        const int cur = it & 1, nxt = cur ^ 1;
        const int kn = (it + 1) * 32;

        // ---- issue chunk i+1 loads (pinned to this region by the barriers)
        a[nxt].h.lo = *(const v8bf*)(parow + kn);
        a[nxt].h.hi = *(const v8bf*)(parow + kn + 8);
#pragma unroll
        for (int j = 0; j < JT; ++j) b[nxt][j] = *(const v16bf*)(pw[j] + kn);

        // ---- WMMA chain on chunk i (depends on previous barrier's acc)
#pragma unroll
        for (int j = 0; j < JT; ++j)
            acc[j] = __builtin_amdgcn_wmma_f32_16x16x32_bf16(
                false, a[cur].v, false, b[cur][j], (short)0, acc[j], false, false);

        acc_barrier<JT>(acc);
    }
    // epilogue: last k chunk
    {
        const int last = (NK - 1) & 1;
#pragma unroll
        for (int j = 0; j < JT; ++j)
            acc[j] = __builtin_amdgcn_wmma_f32_16x16x32_bf16(
                false, a[last].v, false, b[last][j], (short)0, acc[j], false, false);
    }

    // ---- store: C/D layout: elem r -> M = m0 + r + 8*half, N = n0 + (lane&15)
#pragma unroll
    for (int j = 0; j < JT; ++j) {
        const int n0 = nbase + j * 16;
        const float bv = bias[n0 + row16];
#pragma unroll
        for (int r = 0; r < 8; ++r) {
            C[(size_t)(m0 + r + 8 * half) * ldc + n0 + row16] = acc[j][r] + bv;
        }
    }
}

// Host-side dispatch: full 64-wide blocks + one JT-remainder block (N mult of 16).
template <int K>
static void launch_gemm(const bf16_t* A, const bf16_t* W, const float* bias,
                        float* C, int M, int N, int ldc, hipStream_t stream)
{
    const int mblocks = (M + 63) / 64;
    const int nfull = N / 64;
    if (nfull > 0)
        gemm_bf16_wmma<4, K><<<dim3(nfull, mblocks), 128, 0, stream>>>(A, W, bias, C, M, ldc);
    const int rem = (N - nfull * 64) / 16;     // 0..3
    if (rem > 0) {
        const bf16_t* Wt = W + (size_t)nfull * 64 * K;
        const float*  bt = bias + nfull * 64;
        float*        Ct = C + nfull * 64;
        switch (rem) {
        case 1: gemm_bf16_wmma<1, K><<<dim3(1, mblocks), 128, 0, stream>>>(A, Wt, bt, Ct, M, ldc); break;
        case 2: gemm_bf16_wmma<2, K><<<dim3(1, mblocks), 128, 0, stream>>>(A, Wt, bt, Ct, M, ldc); break;
        default: gemm_bf16_wmma<3, K><<<dim3(1, mblocks), 128, 0, stream>>>(A, Wt, bt, Ct, M, ldc); break;
        }
    }
}

// ---------------------------------------------------------------------------
// Elementwise f32 -> bf16 conversion (grid-stride).
// ---------------------------------------------------------------------------
__global__ void f32_to_bf16_k(const float* __restrict__ src, bf16_t* __restrict__ dst,
                              long n)
{
    long i = (long)blockIdx.x * blockDim.x + threadIdx.x;
    const long stride = (long)gridDim.x * blockDim.x;
    for (; i < n; i += stride) dst[i] = (bf16_t)src[i];
}

// ---------------------------------------------------------------------------
// Embedding gather: embs[t, b, :] = bf16(table[captions[b, t], :]), t in 0..30
// ---------------------------------------------------------------------------
__global__ void gather_emb_k(const float* __restrict__ table, const int* __restrict__ captions,
                             bf16_t* __restrict__ embs)
{
    const int t = blockIdx.x / DEC_B;
    const int b = blockIdx.x % DEC_B;
    const int w = captions[b * DEC_T + t];   // uses columns 0..T-2
    const float* src = table + (size_t)w * DEC_EMB;
    bf16_t* dst = embs + ((size_t)t * DEC_B + b) * DEC_EMB;
    for (int i = threadIdx.x; i < DEC_EMB; i += blockDim.x) dst[i] = (bf16_t)src[i];
}

// ---------------------------------------------------------------------------
// h = 0 (both f32 and bf16 copies)
// ---------------------------------------------------------------------------
__global__ void init_h_k(float* __restrict__ h, bf16_t* __restrict__ hbf)
{
    const int i = blockIdx.x * blockDim.x + threadIdx.x;
    if (i < DEC_B * DEC_DEC) { h[i] = 0.0f; hbf[i] = (bf16_t)0.0f; }
}

// ---------------------------------------------------------------------------
// Attention step: e = tanh(att1 + att2) . w_full + b_full; softmax over P;
// context = sum_p alpha_p * enc[b,p,:]; x = [emb_t | context] in bf16.
// One block (256 threads, 8 waves) per batch element.
// ---------------------------------------------------------------------------
__global__ __launch_bounds__(256)
void attention_step_k(const float* __restrict__ att1, const float* __restrict__ att2,
                      const float* __restrict__ w_full, const float* __restrict__ b_full,
                      const float* __restrict__ enc, const bf16_t* __restrict__ emb_t,
                      bf16_t* __restrict__ x)
{
    const int b    = blockIdx.x;
    const int tid  = threadIdx.x;
    const int wave = tid >> 5;
    const int lane = tid & 31;

    __shared__ float s_att2[DEC_ATT];
    __shared__ float s_w[DEC_ATT];
    __shared__ float s_e[DEC_P];
    __shared__ float s_red[8];

    for (int i = tid; i < DEC_ATT; i += 256) {
        s_att2[i] = att2[b * DEC_ATT + i];
        s_w[i]    = w_full[i];
    }
    __syncthreads();

    // ---- scores e[p]
    for (int p = wave; p < DEC_P; p += 8) {
        const float* rowp = att1 + ((size_t)b * DEC_P + p) * DEC_ATT;
        float sum = 0.0f;
#pragma unroll 4
        for (int i = lane; i < DEC_ATT; i += 32)
            sum += tanhf(rowp[i] + s_att2[i]) * s_w[i];
#pragma unroll
        for (int off = 16; off > 0; off >>= 1) sum += __shfl_xor(sum, off, 32);
        if (lane == 0) s_e[p] = sum + b_full[0];
    }
    __syncthreads();

    // ---- softmax max
    float m = -3.0e38f;
    for (int p = tid; p < DEC_P; p += 256) m = fmaxf(m, s_e[p]);
#pragma unroll
    for (int off = 16; off > 0; off >>= 1) m = fmaxf(m, __shfl_xor(m, off, 32));
    if (lane == 0) s_red[wave] = m;
    __syncthreads();
    m = s_red[0];
#pragma unroll
    for (int w = 1; w < 8; ++w) m = fmaxf(m, s_red[w]);
    __syncthreads();

    // ---- exp & sum
    float ssum = 0.0f;
    for (int p = tid; p < DEC_P; p += 256) {
        const float ex = __expf(s_e[p] - m);
        s_e[p] = ex;
        ssum += ex;
    }
#pragma unroll
    for (int off = 16; off > 0; off >>= 1) ssum += __shfl_xor(ssum, off, 32);
    if (lane == 0) s_red[wave] = ssum;
    __syncthreads();
    float tot = 0.0f;
#pragma unroll
    for (int w = 0; w < 8; ++w) tot += s_red[w];
    const float inv = 1.0f / tot;
    __syncthreads();
    for (int p = tid; p < DEC_P; p += 256) s_e[p] *= inv;
    __syncthreads();

    // ---- context = alpha . enc  -> x[:, 512:2560] (bf16)
    for (int d = tid; d < DEC_ENC; d += 256) {
        const float* ep = enc + (size_t)b * DEC_P * DEC_ENC + d;
        float acc = 0.0f;
#pragma unroll 4
        for (int p = 0; p < DEC_P; ++p) acc += s_e[p] * ep[(size_t)p * DEC_ENC];
        x[(size_t)b * XDIM + DEC_EMB + d] = (bf16_t)acc;
    }
    // ---- x[:, 0:512] = emb_t
    for (int d = tid; d < DEC_EMB; d += 256)
        x[(size_t)b * XDIM + d] = emb_t[b * DEC_EMB + d];
}

// ---------------------------------------------------------------------------
// GRU cell update: r,z,n gates (PyTorch order), in-place h update + bf16 copy.
// ---------------------------------------------------------------------------
__global__ void gru_update_k(const float* __restrict__ gi, const float* __restrict__ gh,
                             float* __restrict__ h, bf16_t* __restrict__ hbf)
{
    const int i = blockIdx.x * blockDim.x + threadIdx.x;
    if (i >= DEC_B * DEC_DEC) return;
    const int b = i >> 9;          // /512
    const int d = i & 511;
    const float* gib = gi + (size_t)b * G3;
    const float* ghb = gh + (size_t)b * G3;
    const float r = 1.0f / (1.0f + __expf(-(gib[d] + ghb[d])));
    const float z = 1.0f / (1.0f + __expf(-(gib[DEC_DEC + d] + ghb[DEC_DEC + d])));
    const float n = tanhf(gib[2 * DEC_DEC + d] + r * ghb[2 * DEC_DEC + d]);
    const float hv = h[i];
    const float hn = (1.0f - z) * n + z * hv;
    h[i]   = hn;
    hbf[i] = (bf16_t)hn;
}

// ---------------------------------------------------------------------------
// Second output: caplens - 1 (stored as float into d_out tail)
// ---------------------------------------------------------------------------
__global__ void caplens_out_k(const int* __restrict__ caplens, float* __restrict__ out)
{
    const int b = threadIdx.x;
    if (b < DEC_B) out[b] = (float)(caplens[b] - 1);
}

// ---------------------------------------------------------------------------
extern "C" void kernel_launch(void* const* d_in, const int* in_sizes, int n_in,
                              void* d_out, int out_size, void* d_ws, size_t ws_size,
                              hipStream_t stream)
{
    const float* encoder_out = (const float*)d_in[0];
    const int*   captions    = (const int*)  d_in[1];
    const int*   caplens     = (const int*)  d_in[2];
    const float* emb_table   = (const float*)d_in[3];
    const float* W_enc       = (const float*)d_in[4];
    const float* b_enc       = (const float*)d_in[5];
    const float* W_dec       = (const float*)d_in[6];
    const float* b_dec       = (const float*)d_in[7];
    const float* w_full      = (const float*)d_in[8];
    const float* b_full      = (const float*)d_in[9];
    const float* W_ih        = (const float*)d_in[10];
    const float* b_ih        = (const float*)d_in[11];
    const float* W_hh        = (const float*)d_in[12];
    const float* b_hh        = (const float*)d_in[13];
    const float* W_fc        = (const float*)d_in[14];
    const float* b_fc        = (const float*)d_in[15];
    float* out = (float*)d_out;

    // ---- workspace carve-up (256B aligned)
    char* ws = (char*)d_ws;
    size_t off = 0;
    auto carve = [&](size_t bytes) -> char* {
        char* p = ws + off;
        off = (off + bytes + 255) & ~(size_t)255;
        return p;
    };
    const size_t BP = (size_t)DEC_B * DEC_P;                     // 12544
    bf16_t* enc_bf   = (bf16_t*)carve(BP * DEC_ENC * 2);         // 51.4 MB
    bf16_t* Wenc_bf  = (bf16_t*)carve((size_t)DEC_ATT * DEC_ENC * 2);
    bf16_t* Wdec_bf  = (bf16_t*)carve((size_t)DEC_ATT * DEC_DEC * 2);
    bf16_t* Wih_bf   = (bf16_t*)carve((size_t)G3 * XDIM * 2);
    bf16_t* Whh_bf   = (bf16_t*)carve((size_t)G3 * DEC_DEC * 2);
    bf16_t* Wfc_bf   = (bf16_t*)carve((size_t)DEC_V * DEC_DEC * 2);  // 30.7 MB
    bf16_t* embs_bf  = (bf16_t*)carve((size_t)(DEC_T - 1) * DEC_B * DEC_EMB * 2);
    float*  att1     = (float*) carve(BP * DEC_ATT * 4);         // 25.7 MB
    float*  att2     = (float*) carve((size_t)DEC_B * DEC_ATT * 4);
    bf16_t* x_bf     = (bf16_t*)carve((size_t)DEC_B * XDIM * 2);
    float*  gi       = (float*) carve((size_t)DEC_B * G3 * 4);
    float*  gh       = (float*) carve((size_t)DEC_B * G3 * 4);
    float*  h_f32    = (float*) carve((size_t)DEC_B * DEC_DEC * 4);
    bf16_t* h_bf     = (bf16_t*)carve((size_t)DEC_B * DEC_DEC * 2);

    // ---- one-time conversions to bf16 (L2-resident compact copies)
    auto conv = [&](const float* s, bf16_t* d, long n) {
        int blocks = (int)((n + 255) / 256);
        if (blocks > 4096) blocks = 4096;
        f32_to_bf16_k<<<blocks, 256, 0, stream>>>(s, d, n);
    };
    conv(encoder_out, enc_bf, (long)BP * DEC_ENC);
    conv(W_enc, Wenc_bf, (long)DEC_ATT * DEC_ENC);
    conv(W_dec, Wdec_bf, (long)DEC_ATT * DEC_DEC);
    conv(W_ih,  Wih_bf,  (long)G3 * XDIM);
    conv(W_hh,  Whh_bf,  (long)G3 * DEC_DEC);
    conv(W_fc,  Wfc_bf,  (long)DEC_V * DEC_DEC);

    gather_emb_k<<<(DEC_T - 1) * DEC_B, 256, 0, stream>>>(emb_table, captions, embs_bf);
    init_h_k<<<(DEC_B * DEC_DEC + 255) / 256, 256, 0, stream>>>(h_f32, h_bf);

    // ---- att1 = enc @ W_enc^T + b_enc   [12544, 512], K=2048
    launch_gemm<DEC_ENC>(enc_bf, Wenc_bf, b_enc, att1, (int)BP, DEC_ATT, DEC_ATT, stream);

    // ---- sequential decode loop (31 steps)
    for (int t = 0; t < DEC_T - 1; ++t) {
        // att2 = h @ W_dec^T + b_dec   [64, 512], K=512
        launch_gemm<DEC_DEC>(h_bf, Wdec_bf, b_dec, att2, DEC_B, DEC_ATT, DEC_ATT, stream);

        // attention + context + x assembly
        attention_step_k<<<DEC_B, 256, 0, stream>>>(
            att1, att2, w_full, b_full, encoder_out,
            embs_bf + (size_t)t * DEC_B * DEC_EMB, x_bf);

        // gi = x @ W_ih^T + b_ih   [64, 1536], K=2560
        launch_gemm<XDIM>(x_bf, Wih_bf, b_ih, gi, DEC_B, G3, G3, stream);

        // gh = h @ W_hh^T + b_hh   [64, 1536], K=512
        launch_gemm<DEC_DEC>(h_bf, Whh_bf, b_hh, gh, DEC_B, G3, G3, stream);

        // GRU cell -> h (in place) + bf16 copy
        gru_update_k<<<(DEC_B * DEC_DEC + 255) / 256, 256, 0, stream>>>(gi, gh, h_f32, h_bf);

        // logits_t = h @ W_fc^T + b_fc -> out[:, t, :]   [64, 30000], K=512
        launch_gemm<DEC_DEC>(h_bf, Wfc_bf, b_fc, out + (size_t)t * DEC_V,
                             DEC_B, DEC_V, (DEC_T - 1) * DEC_V, stream);
    }

    // ---- second output: caplens - 1
    caplens_out_k<<<1, 64, 0, stream>>>(caplens, out + (size_t)DEC_B * (DEC_T - 1) * DEC_V);
}
